// TransformerBlock_4930622456199
// MI455X (gfx1250) — matrix operand order repaired and checked by reading, not compiled, not run on hardware
//
#include <hip/hip_runtime.h>
#include <math.h>

// ---------------- types ----------------
typedef __attribute__((ext_vector_type(16))) __bf16 bf16x16;
typedef __attribute__((ext_vector_type(8)))  __bf16 bf16x8;
typedef __attribute__((ext_vector_type(8)))  float  f32x8;
typedef __attribute__((ext_vector_type(4)))  unsigned u32x4;
typedef __attribute__((ext_vector_type(8)))  unsigned u32x8;

#define BB   2
#define SS   2048
#define DD   2048
#define HH   16
#define HDIM 128
#define FFND 22016
#define MTOT (BB * SS)   // 4096 rows

// ---------------- WMMA helpers ----------------
__device__ __forceinline__ f32x8 wmma_bf16(bf16x16 a, bf16x16 b, f32x8 c) {
  return __builtin_amdgcn_wmma_f32_16x16x32_bf16(
      /*neg_a=*/false, a, /*neg_b=*/false, b,
      /*c_mod=*/(short)0, c, /*reuse_a=*/false, /*reuse_b=*/false);
}

// A-matrix 16x32 bf16 fragment: lane holds row (lane&15),
// 16 contiguous K starting at (lane>>4)*16.
__device__ __forceinline__ bf16x16 load_a_frag(const __bf16* rowk0, int hb) {
  bf16x8 lo = *(const bf16x8*)(rowk0 + hb * 16);
  bf16x8 hi = *(const bf16x8*)(rowk0 + hb * 16 + 8);
  return __builtin_shufflevector(lo, hi, 0,1,2,3,4,5,6,7,8,9,10,11,12,13,14,15);
}

// ---------------- gfx1250 async global->LDS copy (ASYNCcnt path) -------------
__device__ __forceinline__ void async_g2l_b128(void* lds_ptr, const void* gptr) {
  // low 32 bits of a generic LDS pointer are the LDS byte offset
  unsigned off = (unsigned)(unsigned long long)lds_ptr;
  asm volatile("global_load_async_to_lds_b128 %0, %1, off"
               :: "v"(off), "v"(gptr) : "memory");
}
__device__ __forceinline__ void wait_async0() {
  asm volatile("s_wait_asynccnt 0" ::: "memory");
}

// ---------------- gfx1250 Tensor Data Mover: 2D tile global->LDS --------------
// D# packing per CDNA5 ISA (group0) / §8.4 (group1).
// tile_d0/tensor_d0 along contiguous axis; strides in data_size (2B) units.
__device__ __forceinline__ void tdm_load_2d_bf16(unsigned lds_off, const void* gbase,
                                                 unsigned tile_d0, unsigned tile_d1,
                                                 unsigned tensor_d0, unsigned tensor_d1,
                                                 unsigned long long d0_stride) {
  unsigned long long ga = (unsigned long long)gbase;
  u32x4 g0;
  g0[0] = 1u;                                            // count=1 (valid), user mode
  g0[1] = lds_off;                                       // lds_addr (bytes)
  g0[2] = (unsigned)(ga & 0xffffffffu);                  // global_addr[31:0]
  g0[3] = (unsigned)((ga >> 32) & 0x01ffffffu) | (2u << 30);  // addr[56:32] | type=2
  u32x8 g1;
  g1[0] = 1u << 16;                                      // data_size=1 (2 bytes)
  g1[1] = (tensor_d0 & 0xffffu) << 16;                   // tensor_dim0[15:0]
  g1[2] = ((tensor_d0 >> 16) & 0xffffu) | ((tensor_d1 & 0xffffu) << 16);
  g1[3] = ((tensor_d1 >> 16) & 0xffffu) | ((tile_d0 & 0xffffu) << 16);  // tile_dim0
  g1[4] = (tile_d1 & 0xffffu);                           // tile_dim1 ; tile_dim2=0
  g1[5] = (unsigned)(d0_stride & 0xffffffffu);           // tensor_dim0_stride lo32
  g1[6] = (unsigned)((d0_stride >> 32) & 0xffffu);       // stride hi16 ; dim1_stride=0
  g1[7] = 0u;
  asm volatile("tensor_load_to_lds %0, %1" :: "s"(g0), "s"(g1) : "memory");
}

// ---------------- fp32 -> bf16 converter (8-wide) ----------------
__global__ void f2bf8_kernel(const float* __restrict__ src, __bf16* __restrict__ dst,
                             size_t n8) {
  size_t i = (size_t)blockIdx.x * blockDim.x + threadIdx.x;
  size_t stride = (size_t)gridDim.x * blockDim.x;
  for (; i < n8; i += stride) {
    f32x8 v = *(const f32x8*)(src + i * 8);
    bf16x8 o;
#pragma unroll
    for (int e = 0; e < 8; e++) o[e] = (__bf16)v[e];
    *(bf16x8*)(dst + i * 8) = o;
  }
}

// ---------------- RMSNorm: out_bf16 = x * g / (||x||_2 + eps) ----------------
__global__ void rmsnorm_bf16_kernel(const float* __restrict__ x,
                                    const float* __restrict__ g,
                                    __bf16* __restrict__ out, int D) {
  int row = blockIdx.x;
  const float* xr = x + (size_t)row * D;
  float ss = 0.f;
  for (int i = threadIdx.x; i < D; i += blockDim.x) { float v = xr[i]; ss += v * v; }
#pragma unroll
  for (int m = 16; m >= 1; m >>= 1) ss += __shfl_xor(ss, m, 32);
  __shared__ float red[8];
  __shared__ float stot;
  int wave = threadIdx.x >> 5;
  if ((threadIdx.x & 31) == 0) red[wave] = ss;
  __syncthreads();
  if (threadIdx.x == 0) {
    float t = 0.f;
    for (int i = 0; i < 8; i++) t += red[i];
    stot = t;
  }
  __syncthreads();
  float scale = 1.0f / (sqrtf(stot) + 1e-5f);
  __bf16* orow = out + (size_t)row * D;
  for (int i = threadIdx.x; i < D; i += blockDim.x)
    orow[i] = (__bf16)(xr[i] * g[i] * scale);
}

// ---------------- Generic GEMM: C[M,N] = A[M,K] * W[N,K]^T (+bias)(+resid) -----
// block: 256 thr = 8 waves in 4(m) x 2(n); block tile 128x128; wave tile 32x64.
template <bool OUT_BF16>
__global__ void gemm_bf16_kernel(const __bf16* __restrict__ A,
                                 const __bf16* __restrict__ W,
                                 const float* __restrict__ bias,
                                 const float* __restrict__ resid,
                                 void* __restrict__ outv,
                                 int M, int N, int K) {
  const int lane = threadIdx.x & 31;
  const int wave = threadIdx.x >> 5;
  const int wm = wave & 3, wn = wave >> 2;
  const int m0 = blockIdx.y * 128 + wm * 32;
  const int n0 = blockIdx.x * 128 + wn * 64;
  const int hb = lane >> 4;
  const int lr = lane & 15;

  f32x8 acc[2][4] = {};
  const __bf16* Ar[2];
  Ar[0] = A + (size_t)(m0 + lr) * K;
  Ar[1] = A + (size_t)(m0 + 16 + lr) * K;
  const __bf16* Wc[4];
#pragma unroll
  for (int ni = 0; ni < 4; ni++) Wc[ni] = W + (size_t)(n0 + ni * 16 + lr) * K;

  for (int k0 = 0; k0 < K; k0 += 32) {
    bf16x16 a0 = load_a_frag(Ar[0] + k0, hb);
    bf16x16 a1 = load_a_frag(Ar[1] + k0, hb);
    bf16x16 b[4];
#pragma unroll
    for (int ni = 0; ni < 4; ni++)
      b[ni] = *(const bf16x16*)(Wc[ni] + k0 + hb * 16);
#pragma unroll
    for (int ni = 0; ni < 4; ni++) {
      acc[0][ni] = wmma_bf16(a0, b[ni], acc[0][ni]);
      acc[1][ni] = wmma_bf16(a1, b[ni], acc[1][ni]);
    }
  }

#pragma unroll
  for (int mi = 0; mi < 2; mi++)
#pragma unroll
    for (int ni = 0; ni < 4; ni++) {
      int n = n0 + ni * 16 + lr;
      float bv = bias ? bias[n] : 0.f;
#pragma unroll
      for (int r = 0; r < 8; r++) {
        int m = m0 + mi * 16 + r + hb * 8;
        size_t idx = (size_t)m * N + n;
        float v = acc[mi][ni][r] + bv;
        if (resid) v += resid[idx];
        if (OUT_BF16) ((__bf16*)outv)[idx] = (__bf16)v;
        else          ((float*)outv)[idx]  = v;
      }
    }
}

// ---------------- Fused FFN w1/w3 GEMM + SwiGLU: out = silu(A w1^T) * (A w3^T) --
__global__ void gemm_ffn13_kernel(const __bf16* __restrict__ A,
                                  const __bf16* __restrict__ W1,
                                  const __bf16* __restrict__ W3,
                                  __bf16* __restrict__ out,
                                  int M, int N, int K) {
  const int lane = threadIdx.x & 31;
  const int wave = threadIdx.x >> 5;
  const int wm = wave & 3, wn = wave >> 2;
  const int m0 = blockIdx.y * 128 + wm * 32;
  const int n0 = blockIdx.x * 64 + wn * 32;
  const int hb = lane >> 4;
  const int lr = lane & 15;

  f32x8 acc1[2][2] = {};
  f32x8 acc3[2][2] = {};
  const __bf16* Ar0 = A + (size_t)(m0 + lr) * K;
  const __bf16* Ar1 = A + (size_t)(m0 + 16 + lr) * K;
  const __bf16* W1c0 = W1 + (size_t)(n0 + lr) * K;
  const __bf16* W1c1 = W1 + (size_t)(n0 + 16 + lr) * K;
  const __bf16* W3c0 = W3 + (size_t)(n0 + lr) * K;
  const __bf16* W3c1 = W3 + (size_t)(n0 + 16 + lr) * K;

  for (int k0 = 0; k0 < K; k0 += 32) {
    bf16x16 a0 = load_a_frag(Ar0 + k0, hb);
    bf16x16 a1 = load_a_frag(Ar1 + k0, hb);
    bf16x16 b10 = *(const bf16x16*)(W1c0 + k0 + hb * 16);
    bf16x16 b11 = *(const bf16x16*)(W1c1 + k0 + hb * 16);
    bf16x16 b30 = *(const bf16x16*)(W3c0 + k0 + hb * 16);
    bf16x16 b31 = *(const bf16x16*)(W3c1 + k0 + hb * 16);
    acc1[0][0] = wmma_bf16(a0, b10, acc1[0][0]);
    acc1[0][1] = wmma_bf16(a0, b11, acc1[0][1]);
    acc1[1][0] = wmma_bf16(a1, b10, acc1[1][0]);
    acc1[1][1] = wmma_bf16(a1, b11, acc1[1][1]);
    acc3[0][0] = wmma_bf16(a0, b30, acc3[0][0]);
    acc3[0][1] = wmma_bf16(a0, b31, acc3[0][1]);
    acc3[1][0] = wmma_bf16(a1, b30, acc3[1][0]);
    acc3[1][1] = wmma_bf16(a1, b31, acc3[1][1]);
  }

#pragma unroll
  for (int mi = 0; mi < 2; mi++)
#pragma unroll
    for (int ni = 0; ni < 2; ni++) {
      int n = n0 + ni * 16 + lr;
#pragma unroll
      for (int r = 0; r < 8; r++) {
        int m = m0 + mi * 16 + r + hb * 8;
        float u = acc1[mi][ni][r];
        float g = acc3[mi][ni][r];
        float silu = u / (1.0f + __expf(-u));
        out[(size_t)m * N + n] = (__bf16)(silu * g);
      }
    }
}

// ---------------- Flash attention (causal, bf16 WMMA, fp32 softmax) -----------
// grid: (S/64, B*H); block 128 thr = 4 waves; wave owns 16 query rows.
// Q block staged by TDM (TENSORcnt); K tile by async-LDS (ASYNCcnt); V manual.
__global__ void flash_attn_kernel(const __bf16* __restrict__ qb,
                                  const __bf16* __restrict__ kb,
                                  const __bf16* __restrict__ vb,
                                  __bf16* __restrict__ ob) {
  const int lane = threadIdx.x & 31;
  const int wave = threadIdx.x >> 5;
  const int hb = lane >> 4;
  const int lr = lane & 15;
  const int q0b = blockIdx.x * 64;
  const int b = (int)blockIdx.y >> 4;     // H == 16
  const int h = (int)blockIdx.y & 15;
  const int q0w = q0b + wave * 16;
  const float scale = 0.08838834764831845f;  // 1/sqrt(128)

  __shared__ __bf16 qlds[64 * HDIM];       // Q block [q][dim]   (TDM target)
  __shared__ __bf16 klds[32 * HDIM];       // K tile  [key][dim] (async target)
  __shared__ __bf16 vtlds[HDIM * 32];      // V^T     [dim][key]
  __shared__ __bf16 plds[4][16 * 32];      // per-wave probs scratch

  // --- TDM: one wave DMA's the whole 64x128 Q block into LDS ---
  if (wave == 0) {
    const __bf16* gq = qb + ((size_t)(b * SS + q0b) * DD) + h * HDIM;
    tdm_load_2d_bf16((unsigned)(unsigned long long)&qlds[0], gq,
                     /*tile d0,d1=*/HDIM, 64,
                     /*tensor d0,d1=*/HDIM, 64,
                     /*d0 stride=*/DD);
    __builtin_amdgcn_s_wait_tensorcnt(0);
  }
  __syncthreads();

  // Q fragments from LDS: 16 rows x 128 dims = 4 k-steps
  bf16x16 qfrag[4];
  {
    const __bf16* qrow = &qlds[(wave * 16 + lr) * HDIM];
#pragma unroll
    for (int kk = 0; kk < 4; kk++) qfrag[kk] = load_a_frag(qrow + kk * 32, hb);
  }

  f32x8 acc[8] = {};
  float mi[8], li[8];
#pragma unroll
  for (int r = 0; r < 8; r++) { mi[r] = -3.0e38f; li[r] = 0.f; }

  const int t = threadIdx.x;
  const int jmax = q0b + 64;  // causal: no key block beyond last query of block
  for (int j0 = 0; j0 < jmax; j0 += 32) {
    // K tile: async global->LDS (ASYNCcnt); V tile: manual transposed staging
#pragma unroll
    for (int i = 0; i < 4; i++) {
      int idx = t + i * 128;        // 0..511
      int row = idx >> 4;           // key 0..31
      int c8  = idx & 15;           // 8-elem chunk 0..15
      size_t srow = (size_t)(b * SS + j0 + row) * DD + h * HDIM + c8 * 8;
      async_g2l_b128(&klds[row * HDIM + c8 * 8], kb + srow);
      bf16x8 vv = *(const bf16x8*)(vb + srow);
#pragma unroll
      for (int e = 0; e < 8; e++) vtlds[(c8 * 8 + e) * 32 + row] = vv[e];
    }
    wait_async0();     // this wave's async copies landed in LDS
    __syncthreads();   // publish all waves' tiles

    if (j0 <= q0w + 15) {  // wave has at least one unmasked key in this tile
      // S = Q * K^T : 16x32 scores as two 16x16 C tiles
      f32x8 sacc[2] = {};
#pragma unroll
      for (int kk = 0; kk < 4; kk++) {
        bf16x16 bk0 = *(const bf16x16*)&klds[lr * HDIM + kk * 32 + hb * 16];
        bf16x16 bk1 = *(const bf16x16*)&klds[(16 + lr) * HDIM + kk * 32 + hb * 16];
        sacc[0] = wmma_bf16(qfrag[kk], bk0, sacc[0]);
        sacc[1] = wmma_bf16(qfrag[kk], bk1, sacc[1]);
      }
      // online softmax over the 32 keys
#pragma unroll
      for (int r = 0; r < 8; r++) {
        int q = q0w + r + hb * 8;
        float s0 = sacc[0][r] * scale + (((j0 + lr)      > q) ? -1e9f : 0.f);
        float s1 = sacc[1][r] * scale + (((j0 + 16 + lr) > q) ? -1e9f : 0.f);
        float mx = fmaxf(s0, s1);
#pragma unroll
        for (int msk = 1; msk < 16; msk <<= 1) mx = fmaxf(mx, __shfl_xor(mx, msk, 32));
        float m2 = fmaxf(mi[r], mx);
        float corr = __expf(mi[r] - m2);
        float p0 = __expf(s0 - m2);
        float p1 = __expf(s1 - m2);
        float ls = p0 + p1;
#pragma unroll
        for (int msk = 1; msk < 16; msk <<= 1) ls += __shfl_xor(ls, msk, 32);
        li[r] = li[r] * corr + ls;
        mi[r] = m2;
#pragma unroll
        for (int nt = 0; nt < 8; nt++) acc[nt][r] *= corr;
        int ql = r + hb * 8;
        plds[wave][ql * 32 + lr]      = (__bf16)p0;
        plds[wave][ql * 32 + 16 + lr] = (__bf16)p1;
      }
      asm volatile("s_wait_dscnt 0" ::: "memory");
      // re-read probs in A-operand layout
      bf16x8 plo = *(const bf16x8*)&plds[wave][lr * 32 + hb * 16];
      bf16x8 phi = *(const bf16x8*)&plds[wave][lr * 32 + hb * 16 + 8];
      bf16x16 pA = __builtin_shufflevector(plo, phi,
          0,1,2,3,4,5,6,7,8,9,10,11,12,13,14,15);
      // O += P * V  (B operand from V^T tile: 16 contiguous keys per lane)
#pragma unroll
      for (int nt = 0; nt < 8; nt++) {
        bf16x16 bv = *(const bf16x16*)&vtlds[(nt * 16 + lr) * 32 + hb * 16];
        acc[nt] = wmma_bf16(pA, bv, acc[nt]);
      }
    }
    __syncthreads();
  }

  // normalize + store bf16
#pragma unroll
  for (int r = 0; r < 8; r++) {
    float inv = 1.0f / li[r];
    int q = q0w + r + hb * 8;
    __bf16* orow = ob + ((size_t)(b * SS + q) * DD) + h * HDIM;
#pragma unroll
    for (int nt = 0; nt < 8; nt++)
      orow[nt * 16 + lr] = (__bf16)(acc[nt][r] * inv);
  }
}

// ---------------- host orchestration ----------------
extern "C" void kernel_launch(void* const* d_in, const int* in_sizes, int n_in,
                              void* d_out, int out_size, void* d_ws, size_t ws_size,
                              hipStream_t stream) {
  (void)in_sizes; (void)n_in; (void)out_size; (void)ws_size;
  const float* x  = (const float*)d_in[0];
  const float* wq = (const float*)d_in[4];
  const float* bq = (const float*)d_in[5];
  const float* wk = (const float*)d_in[6];
  const float* bk = (const float*)d_in[7];
  const float* wv = (const float*)d_in[8];
  const float* bv = (const float*)d_in[9];
  const float* wo = (const float*)d_in[10];
  const float* bo = (const float*)d_in[11];
  const float* w1 = (const float*)d_in[12];
  const float* w2 = (const float*)d_in[13];
  const float* w3 = (const float*)d_in[14];
  const float* g_attn = (const float*)d_in[15];
  const float* g_ffn  = (const float*)d_in[16];
  float* out = (float*)d_out;

  char* p = (char*)d_ws;
  auto alloc = [&](size_t bytes) {
    char* r = p;
    p += (bytes + 255) & ~(size_t)255;
    return r;
  };
  const size_t NDD  = (size_t)DD * DD;      // 4.19M
  const size_t NFD  = (size_t)FFND * DD;    // 45.1M
  const size_t NMD  = (size_t)MTOT * DD;    // 8.39M
  const size_t NMF  = (size_t)MTOT * FFND;  // 90.2M

  __bf16* wq_b = (__bf16*)alloc(NDD * 2);
  __bf16* wk_b = (__bf16*)alloc(NDD * 2);
  __bf16* wv_b = (__bf16*)alloc(NDD * 2);
  __bf16* wo_b = (__bf16*)alloc(NDD * 2);
  __bf16* w1_b = (__bf16*)alloc(NFD * 2);
  __bf16* w2_b = (__bf16*)alloc(NFD * 2);
  __bf16* w3_b = (__bf16*)alloc(NFD * 2);
  __bf16* xn_b = (__bf16*)alloc(NMD * 2);
  __bf16* q_b  = (__bf16*)alloc(NMD * 2);
  __bf16* k_b  = (__bf16*)alloc(NMD * 2);
  __bf16* v_b  = (__bf16*)alloc(NMD * 2);
  __bf16* at_b = (__bf16*)alloc(NMD * 2);
  float*  h_f  = (float*)alloc(NMD * 4);
  __bf16* hn_b = (__bf16*)alloc(NMD * 2);
  __bf16* ff_b = (__bf16*)alloc(NMF * 2);

  // 1) weights -> bf16
  f2bf8_kernel<<<2048, 256, 0, stream>>>(wq, wq_b, NDD / 8);
  f2bf8_kernel<<<2048, 256, 0, stream>>>(wk, wk_b, NDD / 8);
  f2bf8_kernel<<<2048, 256, 0, stream>>>(wv, wv_b, NDD / 8);
  f2bf8_kernel<<<2048, 256, 0, stream>>>(wo, wo_b, NDD / 8);
  f2bf8_kernel<<<4096, 256, 0, stream>>>(w1, w1_b, NFD / 8);
  f2bf8_kernel<<<4096, 256, 0, stream>>>(w2, w2_b, NFD / 8);
  f2bf8_kernel<<<4096, 256, 0, stream>>>(w3, w3_b, NFD / 8);

  // 2) xn = rmsnorm(x, g_attn)
  rmsnorm_bf16_kernel<<<MTOT, 256, 0, stream>>>(x, g_attn, xn_b, DD);

  // 3) Q/K/V projections (bf16 out)
  dim3 gP(DD / 128, MTOT / 128);
  gemm_bf16_kernel<true><<<gP, 256, 0, stream>>>(xn_b, wq_b, bq, nullptr, q_b, MTOT, DD, DD);
  gemm_bf16_kernel<true><<<gP, 256, 0, stream>>>(xn_b, wk_b, bk, nullptr, k_b, MTOT, DD, DD);
  gemm_bf16_kernel<true><<<gP, 256, 0, stream>>>(xn_b, wv_b, bv, nullptr, v_b, MTOT, DD, DD);

  // 4) flash attention
  dim3 gA(SS / 64, BB * HH);
  flash_attn_kernel<<<gA, 128, 0, stream>>>(q_b, k_b, v_b, at_b);

  // 5) h = x + attn @ wo^T + bo  (fp32 out)
  gemm_bf16_kernel<false><<<gP, 256, 0, stream>>>(at_b, wo_b, bo, x, h_f, MTOT, DD, DD);

  // 6) hn = rmsnorm(h, g_ffn)
  rmsnorm_bf16_kernel<<<MTOT, 256, 0, stream>>>(h_f, g_ffn, hn_b, DD);

  // 7) ff = silu(hn @ w1^T) * (hn @ w3^T)   (bf16 out)
  dim3 gF(FFND / 64, MTOT / 128);
  gemm_ffn13_kernel<<<gF, 256, 0, stream>>>(hn_b, w1_b, w3_b, ff_b, MTOT, FFND, DD);

  // 8) out = h + ff @ w2^T  (fp32 out)
  gemm_bf16_kernel<false><<<gP, 256, 0, stream>>>(ff_b, w2_b, nullptr, h_f, out, MTOT, DD, FFND);
}
